// TSPGNNEncoder_10617159156303
// MI455X (gfx1250) — compile-verified, matrix operand order checked by reading, not run on hardware
//
#include <hip/hip_runtime.h>
#include <hip/hip_bf16.h>
#include <math.h>

// ---------------------------------------------------------------------------
// TSP GNN encoder for MI455X (gfx1250, wave32, WMMA).
// All dense GEMMs (K=N=256) run through v_wmma_f32_16x16x32_bf16 with f32
// accumulation. Weights are pre-packed once into the WMMA B-fragment lane
// layout so the inner loop is pure b128 loads + WMMA. Each wave computes a
// 16x64 output strip (4 N-tiles) so every A fragment feeds 4 back-to-back
// WMMAs from the same registers: 2.5 b128 loads per WMMA. Entire working set
// is L2-resident on MI455X (192 MB L2), so throughput is bounded by the
// matrix pipes, not HBM.
// ---------------------------------------------------------------------------

#define V_NODES 3000
#define K_DEG   16
#define E_EDGES 48000
#define H       256
#define TE      128
#define L_LAYERS 12

typedef __bf16 bf16_t;
typedef bf16_t v8bf  __attribute__((ext_vector_type(8)));
typedef bf16_t v16bf __attribute__((ext_vector_type(16)));
typedef float  v8f   __attribute__((ext_vector_type(8)));

__device__ __forceinline__ float wave_sum(float v) {
  #pragma unroll
  for (int off = 16; off > 0; off >>= 1) v += __shfl_xor(v, off, 32);
  return v;
}

// ---------------------------------------------------------------------------
// Pack f32 weight matrices (256x256, row-major KxN) into WMMA B-fragment
// order: dst[((kt*16+nt)*32 + lane)*16 + h] = W[k][n], with
//   k = kt*32 + (lane>>4)*8 + (h>>3)*16 + (h&7),  n = nt*16 + (lane&15)
// matching the CDNA5 16-bit 32x16 B layout. Each GEMM lane then reads its
// whole fragment as one contiguous 32B chunk (two b128 loads).
// ---------------------------------------------------------------------------
__global__ void pack_w(const float* __restrict__ src, bf16_t* __restrict__ dst,
                       int nmats) {
  int total = nmats << 16;
  for (int o = blockIdx.x * blockDim.x + threadIdx.x; o < total;
       o += gridDim.x * blockDim.x) {
    int mat = o >> 16;
    int r   = o & 65535;
    int h   = r & 15;
    int l   = (r >> 4) & 31;
    int nt  = (r >> 9) & 15;
    int kt  = r >> 13;
    int k = kt * 32 + ((l >> 4) << 3) + ((h >> 3) << 4) + (h & 7);
    int n = (nt << 4) + (l & 15);
    dst[o] = (bf16_t)src[(size_t)mat * 65536 + k * 256 + n];
  }
}

// ---------------------------------------------------------------------------
// WMMA GEMM: C[M,256] = A[M,256](bf16) @ Wp(packed bf16) + bias.
//   mode 0: store f32 only
//   mode 1: store f32 + bf16 mirror
//   mode 2: Cf += result (residual), refresh bf16 mirror
// Block = 8 waves covering a 32x256 strip: wave w -> rows (w>>2)*16,
// N-tiles (w&3)*4 .. +3. Per K-step each wave loads one A fragment
// (two b128, straight from row-major bf16 A) and four packed B fragments,
// issuing 4 WMMAs that all reuse the A registers.
// ---------------------------------------------------------------------------
__global__ void __launch_bounds__(256) gemm_wmma(
    const bf16_t* __restrict__ A, const bf16_t* __restrict__ Wp,
    const float* __restrict__ bias, float* __restrict__ Cf,
    bf16_t* __restrict__ Cb, int M, int mode) {
  int lane = threadIdx.x & 31;
  int wave = threadIdx.x >> 5;
  int ntb  = (wave & 3) * 4;                      // first of 4 N-tiles
  int m0   = blockIdx.x * 32 + (wave >> 2) * 16;  // 16-row strip
  int hi   = lane >> 4;
  int lan  = lane & 15;
  int m    = m0 + lan;
  int mc   = m < M ? m : M - 1;   // clamp A loads on ragged last tile
  const bf16_t* arow = A + (size_t)mc * 256;

  v8f acc[4];
  #pragma unroll
  for (int j = 0; j < 4; j++) {
    float bv = bias[(ntb + j) * 16 + lan];
    #pragma unroll
    for (int r = 0; r < 8; r++) acc[j][r] = bv;
  }

  #pragma unroll
  for (int kt = 0; kt < 8; kt++) {
    v8bf a0 = *(const v8bf*)(arow + kt * 32 + hi * 8);
    v8bf a1 = *(const v8bf*)(arow + kt * 32 + 16 + hi * 8);
    v16bf af;
    #pragma unroll
    for (int i = 0; i < 8; i++) { af[i] = a0[i]; af[i + 8] = a1[i]; }
    #pragma unroll
    for (int j = 0; j < 4; j++) {
      v16bf bfr =
          *(const v16bf*)(Wp + ((size_t)(kt * 16 + ntb + j) * 32 + lane) * 16);
      acc[j] = __builtin_amdgcn_wmma_f32_16x16x32_bf16(
          false, af, false, bfr, (short)0, acc[j], false, false);
    }
  }

  #pragma unroll
  for (int j = 0; j < 4; j++) {
    int n = (ntb + j) * 16 + lan;
    #pragma unroll
    for (int r = 0; r < 8; r++) {
      int mr = m0 + hi * 8 + r;
      if (mr < M) {
        size_t idx = (size_t)mr * 256 + n;
        float v = acc[j][r];
        if (mode == 2) v += Cf[idx];
        Cf[idx] = v;
        if (mode != 0) Cb[idx] = (bf16_t)v;
      }
    }
  }
}

// --------------------------- embeddings ------------------------------------
__global__ void embed_nodes(const float* __restrict__ coords,
                            bf16_t* __restrict__ pe) {
  int i = blockIdx.x * blockDim.x + threadIdx.x;
  if (i >= V_NODES * H) return;
  int v = i >> 8, h = i & 255;
  float coord = coords[v * 2 + (h < 128 ? 0 : 1)];
  int j = h & 127;
  float dt  = expf(logf(10000.0f) * (2.0f * (float)(j >> 1) / 128.0f));
  float val = coord * 6.283185307179586f / dt;
  pe[i] = (bf16_t)((j & 1) ? cosf(val) : sinf(val));
}

__global__ void embed_edges(const float* __restrict__ evals,
                            bf16_t* __restrict__ se) {
  int i = blockIdx.x * blockDim.x + threadIdx.x;
  if (i >= E_EDGES * H) return;
  int e = i >> 8, h = i & 255;
  float dt  = expf(logf(10000.0f) * (2.0f * (float)(h >> 1) / 256.0f));
  float val = evals[e] / dt;
  se[i] = (bf16_t)((h & 1) ? cosf(val) : sinf(val));
}

// timestep MLP + per-layer projections, all in one tiny block
__global__ void temb_kernel(const int* __restrict__ t,
                            const float* __restrict__ te1_w, const float* __restrict__ te1_b,
                            const float* __restrict__ te2_w, const float* __restrict__ te2_b,
                            const float* __restrict__ tl_w,  const float* __restrict__ tl_b,
                            float* __restrict__ tproj) {
  __shared__ float s0[H];
  __shared__ float s1[TE];
  __shared__ float s2[TE];
  int tid = threadIdx.x;
  float tf = (float)t[0];
  if (tid < H) {
    int i = (tid < 128) ? tid : tid - 128;
    float a = tf * expf(-logf(10000.0f) * (float)i / 128.0f);
    s0[tid] = (tid < 128) ? cosf(a) : sinf(a);
  }
  __syncthreads();
  if (tid < TE) {
    float acc = te1_b[tid];
    for (int k = 0; k < H; k++) acc += s0[k] * te1_w[k * TE + tid];
    s1[tid] = fmaxf(acc, 0.0f);
  }
  __syncthreads();
  if (tid < TE) {
    float acc = te2_b[tid];
    for (int k = 0; k < TE; k++) acc += s1[k] * te2_w[k * TE + tid];
    s2[tid] = fmaxf(acc, 0.0f);   // relu(temb)
  }
  __syncthreads();
  for (int l = 0; l < L_LAYERS; l++) {
    float acc = tl_b[l * H + tid];
    for (int k = 0; k < TE; k++) acc += s2[k] * tl_w[(l * TE + k) * H + tid];
    tproj[l * H + tid] = acc;
  }
}

__global__ void mask_add(float* __restrict__ ef, bf16_t* __restrict__ eb,
                         const int* __restrict__ mask,
                         const float* __restrict__ memb) {
  int i = blockIdx.x * blockDim.x + threadIdx.x;
  if (i >= E_EDGES * H) return;
  int e = i >> 8, h = i & 255;
  float v = ef[i] + memb[mask[e] * H + h];
  ef[i] = v;
  eb[i] = (bf16_t)v;
}

// ----------------------- per-layer fused stages ----------------------------
// e_new = Ah[col] + Bh[row] + Ce ; gv = sigmoid(e_new) * Vh[col]
// (row = e>>4 since edges are emitted K_DEG per source node, sorted)
__global__ void edge_fuse1(const float* __restrict__ XA,
                           const float* __restrict__ XB,
                           const float* __restrict__ XV,
                           float* __restrict__ enew /*in: Ce, out: e_new*/,
                           float* __restrict__ gv,
                           const int* __restrict__ col) {
  int i = blockIdx.x * blockDim.x + threadIdx.x;
  if (i >= E_EDGES * H) return;
  int e = i >> 8, h = i & 255;
  int c = col[e];
  int r = e >> 4;
  float en = XA[(size_t)c * H + h] + XB[(size_t)r * H + h] + enew[i];
  enew[i] = en;
  float g = 1.0f / (1.0f + expf(-en));
  gv[i] = g * XV[(size_t)c * H + h];
}

// agg(16 consecutive edges) + LayerNorm + ReLU + residual into x
__global__ void __launch_bounds__(256) node_update(
    const float* __restrict__ XU, const float* __restrict__ gv,
    const float* __restrict__ g_, const float* __restrict__ b_,
    float* __restrict__ xf, bf16_t* __restrict__ xb) {
  int wave = threadIdx.x >> 5, lane = threadIdx.x & 31;
  int v = blockIdx.x * 8 + wave;
  if (v >= V_NODES) return;
  int h0 = lane * 8;
  float vals[8];
  #pragma unroll
  for (int j = 0; j < 8; j++) vals[j] = XU[(size_t)v * H + h0 + j];
  for (int ed = 0; ed < K_DEG; ed++) {
    const float* gp = gv + (size_t)(v * K_DEG + ed) * H + h0;
    #pragma unroll
    for (int j = 0; j < 8; j++) vals[j] += gp[j];
  }
  float s = 0.f, q = 0.f;
  #pragma unroll
  for (int j = 0; j < 8; j++) { s += vals[j]; q += vals[j] * vals[j]; }
  s = wave_sum(s); q = wave_sum(q);
  float mean = s * (1.0f / H);
  float var  = q * (1.0f / H) - mean * mean;
  float inv  = rsqrtf(var + 1e-5f);
  #pragma unroll
  for (int j = 0; j < 8; j++) {
    int h = h0 + j;
    float y = fmaxf((vals[j] - mean) * inv * g_[h] + b_[h], 0.0f);
    size_t idx = (size_t)v * H + h;
    float xn = xf[idx] + y;
    xf[idx] = xn;
    xb[idx] = (bf16_t)xn;
  }
}

// relu(LN1(e_new)) + tproj -> silu(LN2(...)) -> bf16 for the plo GEMM
__global__ void __launch_bounds__(256) edge_fuse2(
    const float* __restrict__ enew,
    const float* __restrict__ neg, const float* __restrict__ neb,
    const float* __restrict__ tproj,
    const float* __restrict__ plog, const float* __restrict__ plob,
    bf16_t* __restrict__ sb) {
  int wave = threadIdx.x >> 5, lane = threadIdx.x & 31;
  int e = blockIdx.x * 8 + wave;
  if (e >= E_EDGES) return;
  int h0 = lane * 8;
  const float* row = enew + (size_t)e * H;
  float vals[8];
  #pragma unroll
  for (int j = 0; j < 8; j++) vals[j] = row[h0 + j];

  float s = 0.f, q = 0.f;
  #pragma unroll
  for (int j = 0; j < 8; j++) { s += vals[j]; q += vals[j] * vals[j]; }
  s = wave_sum(s); q = wave_sum(q);
  float mean = s * (1.0f / H);
  float inv  = rsqrtf(q * (1.0f / H) - mean * mean + 1e-5f);
  #pragma unroll
  for (int j = 0; j < 8; j++) {
    int h = h0 + j;
    float y = fmaxf((vals[j] - mean) * inv * neg[h] + neb[h], 0.0f);
    vals[j] = y + tproj[h];
  }
  s = 0.f; q = 0.f;
  #pragma unroll
  for (int j = 0; j < 8; j++) { s += vals[j]; q += vals[j] * vals[j]; }
  s = wave_sum(s); q = wave_sum(q);
  mean = s * (1.0f / H);
  inv  = rsqrtf(q * (1.0f / H) - mean * mean + 1e-5f);
  #pragma unroll
  for (int j = 0; j < 8; j++) {
    int h = h0 + j;
    float z = (vals[j] - mean) * inv * plog[h] + plob[h];
    float sl = z / (1.0f + expf(-z));          // silu
    sb[(size_t)e * H + h] = (bf16_t)sl;
  }
}

// ------------------------------ GroupNorm head -----------------------------
__global__ void gn_zero(float* acc) {
  int i = blockIdx.x * blockDim.x + threadIdx.x;
  if (i < 64) acc[i] = 0.0f;
}

// hierarchical reduction: fixed h per thread (stride multiple of 256),
// LDS combine 8 channels/group, one f32 atomic per group per block
__global__ void __launch_bounds__(256) gn_partial(const float* __restrict__ ef,
                                                  float* __restrict__ acc) {
  __shared__ float ssum[256], ssq[256];
  int tid = threadIdx.x;
  float s = 0.f, q = 0.f;
  for (size_t i = (size_t)blockIdx.x * 256 + tid; i < (size_t)E_EDGES * H;
       i += (size_t)gridDim.x * 256) {
    float v = ef[i];
    s += v; q += v * v;
  }
  ssum[tid] = s; ssq[tid] = q;
  __syncthreads();
  if ((tid & 7) == 0) {
    float as = 0.f, aq = 0.f;
    #pragma unroll
    for (int j = 0; j < 8; j++) { as += ssum[tid + j]; aq += ssq[tid + j]; }
    atomicAdd(&acc[tid >> 3], as);
    atomicAdd(&acc[32 + (tid >> 3)], aq);
  }
}

__global__ void gn_final(float* acc) {
  int g = threadIdx.x;
  if (g < 32) {
    float cnt  = (float)E_EDGES * 8.0f;
    float sum  = acc[g];
    float sq   = acc[32 + g];
    float mean = sum / cnt;
    float var  = sq / cnt - mean * mean;
    acc[g]      = mean;
    acc[32 + g] = rsqrtf(var + 1e-5f);
  }
}

__global__ void __launch_bounds__(256) head_kernel(
    const float* __restrict__ ef, const float* __restrict__ stats,
    const float* __restrict__ gng, const float* __restrict__ gnb,
    const float* __restrict__ ow, const float* __restrict__ ob,
    float* __restrict__ outp) {
  int wave = threadIdx.x >> 5, lane = threadIdx.x & 31;
  int e = blockIdx.x * 8 + wave;
  if (e >= E_EDGES) return;
  int h0 = lane * 8;
  float a0 = 0.f, a1 = 0.f;
  #pragma unroll
  for (int j = 0; j < 8; j++) {
    int h = h0 + j, g = h >> 3;
    float v = (ef[(size_t)e * H + h] - stats[g]) * stats[32 + g];
    v = fmaxf(v * gng[h] + gnb[h], 0.0f);
    a0 += v * ow[h * 2];
    a1 += v * ow[h * 2 + 1];
  }
  a0 = wave_sum(a0); a1 = wave_sum(a1);
  if (lane == 0) {
    outp[e * 2]     = a0 + ob[0];
    outp[e * 2 + 1] = a1 + ob[1];
  }
}

__global__ void copy_x(const float* __restrict__ xf, float* __restrict__ dst) {
  int i = blockIdx.x * blockDim.x + threadIdx.x;
  if (i < V_NODES * H) dst[i] = xf[i];
}

// ---------------------------------------------------------------------------
extern "C" void kernel_launch(void* const* d_in, const int* in_sizes, int n_in,
                              void* d_out, int out_size, void* d_ws, size_t ws_size,
                              hipStream_t stream) {
  const float* nodes_feature = (const float*)d_in[0];
  const float* e_vals        = (const float*)d_in[1];
  const int*   mask          = (const int*)d_in[2];
  const int*   t_            = (const int*)d_in[3];
  const int*   edge_index    = (const int*)d_in[4];
  const float* node_w = (const float*)d_in[5];
  const float* node_b = (const float*)d_in[6];
  const float* edge_w = (const float*)d_in[7];
  const float* edge_b = (const float*)d_in[8];
  const float* te1_w  = (const float*)d_in[9];
  const float* te1_b  = (const float*)d_in[10];
  const float* te2_w  = (const float*)d_in[11];
  const float* te2_b  = (const float*)d_in[12];
  const float* U_w = (const float*)d_in[13];
  const float* U_b = (const float*)d_in[14];
  const float* V_w = (const float*)d_in[15];
  const float* V_b = (const float*)d_in[16];
  const float* A_w = (const float*)d_in[17];
  const float* A_b = (const float*)d_in[18];
  const float* B_w = (const float*)d_in[19];
  const float* B_b = (const float*)d_in[20];
  const float* C_w = (const float*)d_in[21];
  const float* C_b = (const float*)d_in[22];
  const float* nh_g = (const float*)d_in[23];
  const float* nh_b = (const float*)d_in[24];
  const float* ne_g = (const float*)d_in[25];
  const float* ne_b = (const float*)d_in[26];
  const float* tl_w = (const float*)d_in[27];
  const float* tl_b = (const float*)d_in[28];
  const float* plo_g  = (const float*)d_in[29];
  const float* plo_b  = (const float*)d_in[30];
  const float* plo_w  = (const float*)d_in[31];
  const float* plo_b2 = (const float*)d_in[32];
  const float* gn_g  = (const float*)d_in[33];
  const float* gn_b  = (const float*)d_in[34];
  const float* out_w = (const float*)d_in[35];
  const float* out_b = (const float*)d_in[36];
  const float* mask_emb = (const float*)d_in[37];

  const int* col = edge_index + E_EDGES;  // row 1 of (2,E)

  // ---------------- workspace partition (256B aligned) ----------------
  char* base = (char*)d_ws;
  size_t off = 0;
  auto alloc = [&](size_t bytes) -> char* {
    char* p = base + off;
    off += (bytes + 255) & ~(size_t)255;
    return p;
  };
  const size_t MAT = 65536;  // 256x256 elements
  bf16_t* WbfNode = (bf16_t*)alloc(MAT * 2);
  bf16_t* WbfEdge = (bf16_t*)alloc(MAT * 2);
  bf16_t* WbfU    = (bf16_t*)alloc(L_LAYERS * MAT * 2);
  bf16_t* WbfV    = (bf16_t*)alloc(L_LAYERS * MAT * 2);
  bf16_t* WbfA    = (bf16_t*)alloc(L_LAYERS * MAT * 2);
  bf16_t* WbfB    = (bf16_t*)alloc(L_LAYERS * MAT * 2);
  bf16_t* WbfC    = (bf16_t*)alloc(L_LAYERS * MAT * 2);
  bf16_t* WbfPlo  = (bf16_t*)alloc(L_LAYERS * MAT * 2);
  bf16_t* PEbf = (bf16_t*)alloc((size_t)V_NODES * H * 2);
  bf16_t* SEbf = (bf16_t*)alloc((size_t)E_EDGES * H * 2);  // reused as s_bf
  float*  xf   = (float*)alloc((size_t)V_NODES * H * 4);
  bf16_t* xb   = (bf16_t*)alloc((size_t)V_NODES * H * 2);
  float*  ef   = (float*)alloc((size_t)E_EDGES * H * 4);
  bf16_t* eb   = (bf16_t*)alloc((size_t)E_EDGES * H * 2);
  float*  buf1 = (float*)alloc((size_t)E_EDGES * H * 4);   // Ce / e_new
  float*  buf2 = (float*)alloc((size_t)E_EDGES * H * 4);   // gate*Vh
  float*  XU   = (float*)alloc((size_t)V_NODES * H * 4);
  float*  XV   = (float*)alloc((size_t)V_NODES * H * 4);
  float*  XA   = (float*)alloc((size_t)V_NODES * H * 4);
  float*  XB   = (float*)alloc((size_t)V_NODES * H * 4);
  float*  tproj = (float*)alloc((size_t)L_LAYERS * H * 4);
  float*  gnacc = (float*)alloc(64 * 4);

  // ---------------- weight packing (once per call) ----------------
  pack_w<<<256, 256, 0, stream>>>(node_w, WbfNode, 1);
  pack_w<<<256, 256, 0, stream>>>(edge_w, WbfEdge, 1);
  pack_w<<<1536, 256, 0, stream>>>(U_w,   WbfU,   L_LAYERS);
  pack_w<<<1536, 256, 0, stream>>>(V_w,   WbfV,   L_LAYERS);
  pack_w<<<1536, 256, 0, stream>>>(A_w,   WbfA,   L_LAYERS);
  pack_w<<<1536, 256, 0, stream>>>(B_w,   WbfB,   L_LAYERS);
  pack_w<<<1536, 256, 0, stream>>>(C_w,   WbfC,   L_LAYERS);
  pack_w<<<1536, 256, 0, stream>>>(plo_w, WbfPlo, L_LAYERS);

  // ---------------- embeddings ----------------
  embed_nodes<<<(V_NODES * H + 255) / 256, 256, 0, stream>>>(nodes_feature, PEbf);
  embed_edges<<<(E_EDGES * H + 255) / 256, 256, 0, stream>>>(e_vals, SEbf);
  temb_kernel<<<1, 256, 0, stream>>>(t_, te1_w, te1_b, te2_w, te2_b, tl_w, tl_b, tproj);

  const dim3 gemm_blk(256, 1, 1);
  const dim3 grid_node((V_NODES + 31) / 32, 1, 1);
  const dim3 grid_edge(E_EDGES / 32, 1, 1);

  // x = PE @ node_w + node_b ; e = SE @ edge_w + edge_b (+ mask_emb[mask])
  gemm_wmma<<<grid_node, gemm_blk, 0, stream>>>(PEbf, WbfNode, node_b, xf, xb, V_NODES, 1);
  gemm_wmma<<<grid_edge, gemm_blk, 0, stream>>>(SEbf, WbfEdge, edge_b, ef, eb, E_EDGES, 1);
  mask_add<<<(E_EDGES * H + 255) / 256, 256, 0, stream>>>(ef, eb, mask, mask_emb);

  // ---------------- 12 gated-GCN layers ----------------
  for (int l = 0; l < L_LAYERS; l++) {
    const size_t wo = (size_t)l * MAT;
    gemm_wmma<<<grid_node, gemm_blk, 0, stream>>>(xb, WbfU + wo, U_b + l * H, XU, (bf16_t*)nullptr, V_NODES, 0);
    gemm_wmma<<<grid_node, gemm_blk, 0, stream>>>(xb, WbfV + wo, V_b + l * H, XV, (bf16_t*)nullptr, V_NODES, 0);
    gemm_wmma<<<grid_node, gemm_blk, 0, stream>>>(xb, WbfA + wo, A_b + l * H, XA, (bf16_t*)nullptr, V_NODES, 0);
    gemm_wmma<<<grid_node, gemm_blk, 0, stream>>>(xb, WbfB + wo, B_b + l * H, XB, (bf16_t*)nullptr, V_NODES, 0);
    gemm_wmma<<<grid_edge, gemm_blk, 0, stream>>>(eb, WbfC + wo, C_b + l * H, buf1, (bf16_t*)nullptr, E_EDGES, 0);

    edge_fuse1<<<(E_EDGES * H + 255) / 256, 256, 0, stream>>>(XA, XB, XV, buf1, buf2, col);
    node_update<<<(V_NODES + 7) / 8, 256, 0, stream>>>(XU, buf2, nh_g + l * H, nh_b + l * H, xf, xb);
    edge_fuse2<<<E_EDGES / 8, 256, 0, stream>>>(buf1, ne_g + l * H, ne_b + l * H,
                                                tproj + l * H, plo_g + l * H, plo_b + l * H, SEbf);
    // e += silu(...) @ plo_w + plo_b2   (mode 2: residual accumulate)
    gemm_wmma<<<grid_edge, gemm_blk, 0, stream>>>(SEbf, WbfPlo + wo, plo_b2 + l * H, ef, eb, E_EDGES, 2);
  }

  // ---------------- GroupNorm(32) + ReLU + 1x1 conv head ----------------
  gn_zero<<<1, 64, 0, stream>>>(gnacc);
  gn_partial<<<1024, 256, 0, stream>>>(ef, gnacc);
  gn_final<<<1, 32, 0, stream>>>(gnacc);
  float* out_f = (float*)d_out;
  head_kernel<<<E_EDGES / 8, 256, 0, stream>>>(ef, gnacc, gn_g, gn_b, out_w, out_b,
                                               out_f + (size_t)V_NODES * H);
  copy_x<<<(V_NODES * H + 255) / 256, 256, 0, stream>>>(xf, out_f);
}
